// PrimitiveAnsweringModel_85048942396226
// MI455X (gfx1250) — matrix operand order repaired and checked by reading, not compiled, not run on hardware
//
#include <hip/hip_runtime.h>
#include <math.h>

// ---------------------------------------------------------------------------
// Model constants
// ---------------------------------------------------------------------------
#define Dm     384
#define Hh     6
#define HD     64
#define NL     12
#define GDn    2
#define DFF    1536
#define Gg     512
#define SG     32
#define NQ     256
#define NA     256
#define Bb     4
#define AV     640
#define PL     772          // prompt length
#define TOT    1028         // total sequence

// ---------------------------------------------------------------------------
// Types for WMMA
// ---------------------------------------------------------------------------
typedef __bf16 bf16_t;
typedef __attribute__((ext_vector_type(16))) bf16_t v16bf;
typedef __attribute__((ext_vector_type(8)))  float  v8f;

__device__ __forceinline__ bf16_t f2bf(float f) {
  unsigned u = __builtin_bit_cast(unsigned, f);
  unsigned r = (u + 0x7FFFu + ((u >> 16) & 1u)) >> 16;   // round-to-nearest-even
  unsigned short h = (unsigned short)r;
  return __builtin_bit_cast(bf16_t, h);
}

// gelu(x) = x * sigmoid(2*k*(x + 0.044715 x^3)); branch-free (v_exp + v_rcp)
__device__ __forceinline__ float gelu_f(float x) {
  const float k2 = 1.5957691216057308f;  // 2*sqrt(2/pi)
  float a = k2 * (x + 0.044715f * x * x * x);
  return x / (1.f + __expf(-a));
}

// ---------------------------------------------------------------------------
// Operand packing into WMMA fragment order (bf16).
//
// A fragment layout (16x32 per tile, contiguous k order used here):
//   lane = (m%16) + 16*hv ;  lane's 16 elems: k%32 = hv*16 + i
// Packed: Ap[((tileM*ksteps + kstep)*32 + lane)*16 + i]
// ---------------------------------------------------------------------------
__global__ __launch_bounds__(256)
void k_packA(const float* __restrict__ A, bf16_t* __restrict__ Ap, int K)
{
  const int ksteps = K >> 5;
  const int tileM  = blockIdx.y;
  const int idx    = blockIdx.x * 256 + threadIdx.x;   // 0 .. ksteps*512-1
  const int i      = idx & 15;
  const int lane   = (idx >> 4) & 31;
  const int kstep  = idx >> 9;
  const int m      = tileM * 16 + (lane & 15);
  const int hv     = lane >> 4;
  const int k      = kstep * 32 + hv * 16 + i;
  Ap[(size_t)tileM * (ksteps * 512) + idx] = f2bf(A[(size_t)m * K + k]);
}

// B fragment layout (32x16 per tile):
//   lane = (n%16) + 16*hv ; lane's 16 elems: k%32 = hv*16 + i
// Packed: Wp[((tileN*ksteps + kstep)*32 + lane)*16 + i]
__global__ __launch_bounds__(256)
void k_packW(const float* __restrict__ W, bf16_t* __restrict__ Wp, int K, int N)
{
  const int ksteps = K >> 5;
  const int tileN  = blockIdx.y;
  const int idx    = blockIdx.x * 256 + threadIdx.x;
  const int i      = idx & 15;
  const int lane   = (idx >> 4) & 31;
  const int kstep  = idx >> 9;
  const int n      = tileN * 16 + (lane & 15);
  const int k      = kstep * 32 + (lane >> 4) * 16 + i;
  Wp[(size_t)tileN * (ksteps * 512) + idx] = f2bf(W[(size_t)k * N + n]);
}

// ---------------------------------------------------------------------------
// WMMA GEMM on packed bf16 operands.
// C[M,N] = (R? R:0) + act(A @ W + bias) ; f32 accumulate.
// Each wave: one M tile x two N tiles (A fragment reused across 2 WMMAs).
// Block = 4 waves -> 128 columns. Requires M%16==0, N%128==0, K%32==0.
// ---------------------------------------------------------------------------
__global__ __launch_bounds__(128)
void k_gemm_bf16(const bf16_t* __restrict__ Ap, const bf16_t* __restrict__ Wp,
                 const float* __restrict__ bias, const float* __restrict__ R,
                 float* __restrict__ C, int M, int N, int K, int act)
{
  const int wave  = threadIdx.x >> 5;
  const int lane  = threadIdx.x & 31;
  const int tn0   = (blockIdx.x * 4 + wave) * 2;   // first of two N tiles
  const int tileM = blockIdx.y;
  const int ksteps = K >> 5;
  const int l15 = lane & 15;
  const int hv  = lane >> 4;
  if (tn0 * 16 >= N) return;

  const bf16_t* ap  = Ap + (((size_t)tileM * ksteps) * 32 + lane) * 16;
  const bf16_t* wp0 = Wp + (((size_t)tn0 * ksteps) * 32 + lane) * 16;
  const bf16_t* wp1 = wp0 + (size_t)ksteps * 512;

  v8f c0 = {0.f,0.f,0.f,0.f,0.f,0.f,0.f,0.f};
  v8f c1 = {0.f,0.f,0.f,0.f,0.f,0.f,0.f,0.f};

  for (int ks = 0; ks < ksteps; ++ks) {
    v16bf a  = *(const v16bf*)ap;
    v16bf b0 = *(const v16bf*)wp0;
    v16bf b1 = *(const v16bf*)wp1;
    __builtin_prefetch(ap  + 2 * 512, 0, 1);
    __builtin_prefetch(wp0 + 2 * 512, 0, 1);
    __builtin_prefetch(wp1 + 2 * 512, 0, 1);
    c0 = __builtin_amdgcn_wmma_f32_16x16x32_bf16(false, a, false, b0,
                                                 (short)0, c0, false, false);
    c1 = __builtin_amdgcn_wmma_f32_16x16x32_bf16(false, a, false, b1,
                                                 (short)0, c1, false, false);
    ap += 512; wp0 += 512; wp1 += 512;
  }

  const int n0 = tn0 * 16 + l15;
  const int n1 = n0 + 16;
#pragma unroll
  for (int r = 0; r < 8; ++r) {
    const int mr = tileM * 16 + hv * 8 + r;   // C: VGPR r -> M=r / M=8+r by lane half
    {
      float t = c0[r] + bias[n0];
      if (act) t = gelu_f(t);
      size_t off = (size_t)mr * N + n0;
      if (R) t += R[off];
      C[off] = t;
    }
    {
      float t = c1[r] + bias[n1];
      if (act) t = gelu_f(t);
      size_t off = (size_t)mr * N + n1;
      if (R) t += R[off];
      C[off] = t;
    }
  }
}

// ---------------------------------------------------------------------------
// LayerNorm over rows of 384. One wave per row (12 elems/lane), 8 rows/block.
// ---------------------------------------------------------------------------
__global__ __launch_bounds__(256)
void k_layernorm(const float* __restrict__ x, float* __restrict__ y,
                 const float* __restrict__ g, const float* __restrict__ bta,
                 int rows)
{
  int wave = threadIdx.x >> 5, lane = threadIdx.x & 31;
  int row = blockIdx.x * 8 + wave;
  if (row >= rows) return;
  const float* xr = x + (size_t)row * Dm;
  float v[12], s = 0.f, ss = 0.f;
#pragma unroll
  for (int i = 0; i < 12; ++i) { float t = xr[lane + 32*i]; v[i] = t; s += t; ss += t*t; }
  for (int o = 16; o > 0; o >>= 1) { s += __shfl_xor(s, o, 32); ss += __shfl_xor(ss, o, 32); }
  float mean = s * (1.f/Dm);
  float var  = ss * (1.f/Dm) - mean*mean;
  float rs = rsqrtf(var + 1e-5f);
  float* yr = y + (size_t)row * Dm;
#pragma unroll
  for (int i = 0; i < 12; ++i) { int d = lane + 32*i; yr[d] = (v[i]-mean)*rs*g[d] + bta[d]; }
}

// ---------------------------------------------------------------------------
// Attention. qkv: (B,S,1152) [q|k|v], out: (B,S,384).
// Block = 8 waves, each wave one query row of one (b,h). LDS-staged K/V chunks.
// mode 0: backbone mask;  mode 1: causal.
// ---------------------------------------------------------------------------
#define AT_ROWS 8
#define MAXCH   (17*64)   // >= ceil(1028/64)*64
__global__ __launch_bounds__(256)
void k_attn(const float* __restrict__ qkv, float* __restrict__ out, int S, int mode)
{
  __shared__ float qsh[AT_ROWS][HD];
  __shared__ float kv[64][HD + 1];
  __shared__ float pbuf[AT_ROWS][MAXCH + 4];

  const int b = blockIdx.z, h = blockIdx.y;
  const int rowbase = blockIdx.x * AT_ROWS;
  const int wave = threadIdx.x >> 5, lane = threadIdx.x & 31;
  const int tid = threadIdx.x;

  for (int i = tid; i < AT_ROWS * HD; i += blockDim.x) {
    int r = i >> 6, d = i & 63;
    int qi = rowbase + r;
    qsh[r][d] = (qi < S) ? qkv[((size_t)b * S + qi) * 1152 + h * HD + d] : 0.f;
  }
  __syncthreads();

  const int qi = rowbase + wave;
  const bool active = (qi < S);
  const int nch = (S + 63) >> 6;

  // Phase A: scores
  for (int cb = 0; cb < nch; ++cb) {
    for (int i = tid; i < 64 * HD; i += blockDim.x) {
      int j = i >> 6, d = i & 63;
      int kj = cb * 64 + j;
      kv[j][d] = (kj < S) ? qkv[((size_t)b * S + kj) * 1152 + Dm + h * HD + d] : 0.f;
    }
    __syncthreads();
    if (active) {
#pragma unroll
      for (int t = 0; t < 2; ++t) {
        int jj = lane + t * 32;
        int kj = cb * 64 + jj;
        float sc = -__builtin_inff();
        if (kj < S) {
          float acc = 0.f;
#pragma unroll
          for (int d = 0; d < HD; ++d) acc += qsh[wave][d] * kv[jj][d];
          sc = acc * 0.125f;   // 1/sqrt(64)
          bool masked = (mode == 0) ? ((kj >= PL) && ((qi < PL) || (kj > qi)))
                                    : (kj > qi);
          if (masked) sc = -1e9f;
        }
        pbuf[wave][cb * 64 + jj] = sc;
      }
    }
    __syncthreads();
  }

  // Softmax (row private to wave)
  float m = -__builtin_inff(), s = 0.f;
  if (active) for (int j = lane; j < nch * 64; j += 32) m = fmaxf(m, pbuf[wave][j]);
  for (int o = 16; o > 0; o >>= 1) m = fmaxf(m, __shfl_xor(m, o, 32));
  if (active) for (int j = lane; j < nch * 64; j += 32) s += __expf(pbuf[wave][j] - m);
  for (int o = 16; o > 0; o >>= 1) s += __shfl_xor(s, o, 32);
  float inv = (s > 0.f) ? 1.f / s : 0.f;
  if (active) for (int j = lane; j < nch * 64; j += 32)
    pbuf[wave][j] = __expf(pbuf[wave][j] - m) * inv;

  // Phase C: O = P @ V ; lane owns dims 2*lane, 2*lane+1
  float a0 = 0.f, a1 = 0.f;
  const int d0 = lane * 2, d1 = lane * 2 + 1;
  for (int cb = 0; cb < nch; ++cb) {
    __syncthreads();
    for (int i = tid; i < 64 * HD; i += blockDim.x) {
      int j = i >> 6, d = i & 63;
      int kj = cb * 64 + j;
      kv[j][d] = (kj < S) ? qkv[((size_t)b * S + kj) * 1152 + 2 * Dm + h * HD + d] : 0.f;
    }
    __syncthreads();
    if (active) {
#pragma unroll 4
      for (int jj = 0; jj < 64; ++jj) {
        float p = pbuf[wave][cb * 64 + jj];
        a0 += p * kv[jj][d0];
        a1 += p * kv[jj][d1];
      }
    }
  }
  if (active) {
    size_t o = ((size_t)b * S + qi) * Dm + h * HD;
    out[o + d0] = a0;
    out[o + d1] = a1;
  }
}

// ---------------------------------------------------------------------------
// Patch embedding: one block per (b,group). 32 points, pe1 3->128->256,
// group-max concat, pe2 512->512->384, max-pool -> token.
// ---------------------------------------------------------------------------
__global__ __launch_bounds__(128)
void k_patch(const float* __restrict__ xyz,
             const float* __restrict__ w1a, const float* __restrict__ b1a,
             const float* __restrict__ w2a, const float* __restrict__ b2a,
             const float* __restrict__ w1b, const float* __restrict__ b1b,
             const float* __restrict__ w2b, const float* __restrict__ b2b,
             float* __restrict__ tok_out, float* __restrict__ ctr_out)
{
  __shared__ float pts[SG][3];
  __shared__ float ctr[3];
  __shared__ float f1[128];
  __shared__ float fbuf[SG][256];
  __shared__ float gmaxs[256];
  __shared__ float g1[512];
  __shared__ float tokm[Dm];

  const int bg = blockIdx.x;
  const int tid = threadIdx.x;

  if (tid < SG * 3) ((float*)pts)[tid] = xyz[(size_t)bg * SG * 3 + tid];
  __syncthreads();
  if (tid < 3) {
    float s = 0.f;
    for (int p = 0; p < SG; ++p) s += pts[p][tid];
    s *= (1.f / SG);
    ctr[tid] = s;
    ctr_out[(size_t)bg * 3 + tid] = s;
  }
  for (int d = tid; d < Dm; d += 128) tokm[d] = -__builtin_inff();
  __syncthreads();

  for (int p = 0; p < SG; ++p) {
    float l0 = pts[p][0] - ctr[0], l1 = pts[p][1] - ctr[1], l2 = pts[p][2] - ctr[2];
    f1[tid] = gelu_f(b1a[tid] + l0 * w1a[tid] + l1 * w1a[128 + tid] + l2 * w1a[256 + tid]);
    __syncthreads();
    for (int d = tid; d < 256; d += 128) {
      float acc = b2a[d];
      for (int j = 0; j < 128; ++j) acc += f1[j] * w2a[(size_t)j * 256 + d];
      fbuf[p][d] = acc;
    }
    __syncthreads();
  }
  for (int d = tid; d < 256; d += 128) {
    float mx = fbuf[0][d];
    for (int p = 1; p < SG; ++p) mx = fmaxf(mx, fbuf[p][d]);
    gmaxs[d] = mx;
  }
  __syncthreads();
  for (int p = 0; p < SG; ++p) {
    for (int d = tid; d < 512; d += 128) {
      float acc = b1b[d];
      for (int j = 0; j < 256; ++j) acc += fbuf[p][j] * w1b[(size_t)j * 512 + d];
      for (int j = 0; j < 256; ++j) acc += gmaxs[j] * w1b[(size_t)(256 + j) * 512 + d];
      g1[d] = gelu_f(acc);
    }
    __syncthreads();
    for (int d = tid; d < Dm; d += 128) {
      float acc = b2b[d];
      for (int j = 0; j < 512; ++j) acc += g1[j] * w2b[(size_t)j * Dm + d];
      tokm[d] = fmaxf(tokm[d], acc);
    }
    __syncthreads();
  }
  for (int d = tid; d < Dm; d += 128) tok_out[(size_t)bg * Dm + d] = tokm[d];
}

// ---------------------------------------------------------------------------
// Small MLP 3 -> 384(gelu) -> 384 per token; accum: += vs =
// ---------------------------------------------------------------------------
__global__ __launch_bounds__(128)
void k_mlp3(const float* __restrict__ in, const float* __restrict__ w1,
            const float* __restrict__ b1, const float* __restrict__ w2,
            const float* __restrict__ b2, float* __restrict__ out, int accum)
{
  __shared__ float xin[3];
  __shared__ float hbuf[Dm];
  const int t = blockIdx.x;
  const int tid = threadIdx.x;
  if (tid < 3) xin[tid] = in[(size_t)t * 3 + tid];
  __syncthreads();
  for (int d = tid; d < Dm; d += 128)
    hbuf[d] = gelu_f(b1[d] + xin[0]*w1[d] + xin[1]*w1[Dm + d] + xin[2]*w1[2*Dm + d]);
  __syncthreads();
  for (int d = tid; d < Dm; d += 128) {
    float acc = b2[d];
    for (int j = 0; j < Dm; ++j) acc += hbuf[j] * w2[(size_t)j * Dm + d];
    if (accum) out[(size_t)t * Dm + d] += acc;
    else       out[(size_t)t * Dm + d]  = acc;
  }
}

// ---------------------------------------------------------------------------
// q_tok = pos + type_emb[qry_type]
// ---------------------------------------------------------------------------
__global__ void k_qtok(const float* __restrict__ pos, const float* __restrict__ type_emb,
                       const int* __restrict__ qtype, float* __restrict__ out, int n)
{
  for (int i = blockIdx.x * blockDim.x + threadIdx.x; i < n; i += gridDim.x * blockDim.x) {
    int d = i % Dm;
    int q = (i / Dm) % NQ;
    int b = i / (Dm * NQ);
    out[i] = pos[i] + type_emb[qtype[b * NQ + q] * Dm + d];
  }
}

// ---------------------------------------------------------------------------
// Sequence assembly -> d_out.seq and ws.x
// ---------------------------------------------------------------------------
__global__ void k_seq(const float* __restrict__ ctx_tok, const float* __restrict__ qtok,
                      const float* __restrict__ pos,
                      const int* __restrict__ qtype, const int* __restrict__ acode,
                      const float* __restrict__ bos, const float* __restrict__ sepcq,
                      const float* __restrict__ sepa, const float* __restrict__ ansbos,
                      const float* __restrict__ type_emb, const float* __restrict__ ans_emb,
                      float* __restrict__ seq_out, float* __restrict__ x_out, int n)
{
  for (int i = blockIdx.x * blockDim.x + threadIdx.x; i < n; i += gridDim.x * blockDim.x) {
    int d = i % Dm;
    int t = (i / Dm) % TOT;
    int b = i / (Dm * TOT);
    float v;
    if (t == 0)            v = bos[d];
    else if (t <= Gg)      v = ctx_tok[((size_t)b * Gg + (t - 1)) * Dm + d];
    else if (t == Gg + 1)  v = sepcq[d];
    else if (t == Gg + 2)  v = type_emb[qtype[b * NQ] * Dm + d];
    else if (t <= Gg + 2 + NQ) v = qtok[((size_t)b * NQ + (t - (Gg + 3))) * Dm + d];
    else if (t == PL - 1)  v = sepa[d];
    else {
      int j = t - PL;
      float e = (j == 0) ? ansbos[d] : ans_emb[acode[b * NA + (j - 1)] * Dm + d];
      v = e + pos[((size_t)b * NA + j) * Dm + d];
    }
    seq_out[i] = v;
    x_out[i]   = v;
  }
}

__global__ void k_mask(float* __restrict__ mask, int n)
{
  for (int i = blockIdx.x * blockDim.x + threadIdx.x; i < n; i += gridDim.x * blockDim.x) {
    int j = i % TOT, r = i / TOT;
    bool m = (j >= PL) && ((r < PL) || (j > r));
    mask[i] = m ? 1.f : 0.f;
  }
}

__global__ void k_slice(const float* __restrict__ h, float* __restrict__ gx, int n)
{
  for (int i = blockIdx.x * blockDim.x + threadIdx.x; i < n; i += gridDim.x * blockDim.x) {
    int d = i % Dm;
    int j = (i / Dm) % NA;
    int b = i / (Dm * NA);
    gx[i] = h[((size_t)b * TOT + PL + j) * Dm + d];
  }
}

// ---------------------------------------------------------------------------
// Host orchestration
// ---------------------------------------------------------------------------
extern "C" void kernel_launch(void* const* d_in, const int* in_sizes, int n_in,
                              void* d_out, int out_size, void* d_ws, size_t ws_size,
                              hipStream_t stream)
{
  (void)in_sizes; (void)out_size; (void)ws_size;
  if (n_in < 202) return;

  auto P = [&](int i) -> const float* { return (const float*)d_in[i]; };
  const float* ctx_xyz = P(0);
  const float* qry_xyz = P(1);
  const int*   qtype   = (const int*)d_in[2];
  const int*   acode   = (const int*)d_in[3];
  // params (insertion order flatten):
  // 4..7 pe1, 8..11 pe2, 12..15 center_pos, 16..19 query_embed,
  // 20 type_emb, 21 ans_emb, 22 bos, 23 sep_cq, 24 sep_a, 25 ans_bos,
  // 26..169 backbone (12 x [ln1_g,ln1_b,qkv_w,qkv_b,proj_w,proj_b,ln2_g,ln2_b,fc1_w,fc1_b,fc2_w,fc2_b])
  // 170,171 backbone_ln; 172..195 generator; 196,197 generator_ln; 198..201 head

  // workspace layout (floats)
  float* wsf    = (float*)d_ws;
  float* xbuf   = wsf;                      // B*TOT*D   = 1,579,008
  float* lnbuf  = xbuf   + 1579008;
  float* qkvbuf = lnbuf  + 1579008;         // 4,737,024
  float* attbuf = qkvbuf + 4737024;
  float* hidbuf = attbuf + 1579008;         // 6,316,032
  float* posbuf = hidbuf + 6316032;         // 393,216
  bf16_t* apack = (bf16_t*)(posbuf + 393216);        // 6,316,032 bf16
  bf16_t* wpack = (bf16_t*)(posbuf + 393216 + 3158016); // 589,824 bf16

  // output layout (floats, return order)
  float* out      = (float*)d_out;
  float* o_logits = out;                    // B*NA*AV  = 655,360
  float* o_h      = o_logits + 655360;      // 1,579,008
  float* o_ctx    = o_h      + 1579008;     // 786,432
  float* o_ctr    = o_ctx    + 786432;      // 6,144
  float* o_qtok   = o_ctr    + 6144;        // 393,216
  float* o_g      = o_qtok   + 393216;      // 393,216
  float* o_seq    = o_g      + 393216;      // 1,579,008
  float* o_mask   = o_seq    + 1579008;     // 1,056,784

  // pack + GEMM helper
  auto gemm = [&](const float* A, const float* W, const float* bias, const float* R,
                  float* C, int M, int N, int K, int act) {
    int ksteps = K >> 5;
    k_packA<<<dim3(ksteps * 2, M / 16), 256, 0, stream>>>(A, apack, K);
    k_packW<<<dim3(ksteps * 2, N / 16), 256, 0, stream>>>(W, wpack, K, N);
    k_gemm_bf16<<<dim3(N / 128, M / 16), 128, 0, stream>>>(apack, wpack, bias, R, C,
                                                           M, N, K, act);
  };

  // 1) patch embedding -> ctx_tok (pre pos-add), centers
  k_patch<<<Bb * Gg, 128, 0, stream>>>(ctx_xyz, P(4), P(5), P(6), P(7),
                                       P(8), P(9), P(10), P(11), o_ctx, o_ctr);
  // 2) ctx_tok += center_pos MLP(centers)
  k_mlp3<<<Bb * Gg, 128, 0, stream>>>(o_ctr, P(12), P(13), P(14), P(15), o_ctx, 1);
  // 3) pos = query_embed MLP(qry_xyz)
  k_mlp3<<<Bb * NQ, 128, 0, stream>>>(qry_xyz, P(16), P(17), P(18), P(19), posbuf, 0);
  // 4) q_tok = pos + type_emb[qry_type]
  {
    int n = Bb * NQ * Dm;
    k_qtok<<<(n + 255) / 256, 256, 0, stream>>>(posbuf, P(20), qtype, o_qtok, n);
  }
  // 5) sequence -> seq output + x
  {
    int n = Bb * TOT * Dm;
    k_seq<<<(n + 255) / 256, 256, 0, stream>>>(o_ctx, o_qtok, posbuf, qtype, acode,
                                               P(22), P(23), P(24), P(25), P(20), P(21),
                                               o_seq, xbuf, n);
  }
  // 6) mask output
  {
    int n = TOT * TOT;
    k_mask<<<(n + 255) / 256, 256, 0, stream>>>(o_mask, n);
  }

  const int M  = Bb * TOT;   // 4112
  const int Mg = Bb * NA;    // 1024

  // 7) backbone: 12 layers
  for (int l = 0; l < NL; ++l) {
    int bi = 26 + 12 * l;
    k_layernorm<<<(M + 7) / 8, 256, 0, stream>>>(xbuf, lnbuf, P(bi + 0), P(bi + 1), M);
    gemm(lnbuf, P(bi + 2), P(bi + 3), nullptr, qkvbuf, M, 3 * Dm, Dm, 0);
    k_attn<<<dim3((TOT + AT_ROWS - 1) / AT_ROWS, Hh, Bb), 256, 0, stream>>>(
        qkvbuf, attbuf, TOT, 0);
    gemm(attbuf, P(bi + 4), P(bi + 5), xbuf, xbuf, M, Dm, Dm, 0);
    k_layernorm<<<(M + 7) / 8, 256, 0, stream>>>(xbuf, lnbuf, P(bi + 6), P(bi + 7), M);
    gemm(lnbuf, P(bi + 8), P(bi + 9), nullptr, hidbuf, M, DFF, Dm, 1);
    gemm(hidbuf, P(bi + 10), P(bi + 11), xbuf, xbuf, M, Dm, DFF, 0);
  }
  // 8) backbone final LN -> h
  k_layernorm<<<(M + 7) / 8, 256, 0, stream>>>(xbuf, o_h, P(170), P(171), M);

  // 9) h_ans slice -> xbuf (reused for generator activations)
  {
    int n = Bb * NA * Dm;
    k_slice<<<(n + 255) / 256, 256, 0, stream>>>(o_h, xbuf, n);
  }

  // 10) generator: 2 layers (causal, S=256)
  for (int l = 0; l < GDn; ++l) {
    int bi = 172 + 12 * l;
    k_layernorm<<<(Mg + 7) / 8, 256, 0, stream>>>(xbuf, lnbuf, P(bi + 0), P(bi + 1), Mg);
    gemm(lnbuf, P(bi + 2), P(bi + 3), nullptr, qkvbuf, Mg, 3 * Dm, Dm, 0);
    k_attn<<<dim3((NA + AT_ROWS - 1) / AT_ROWS, Hh, Bb), 256, 0, stream>>>(
        qkvbuf, attbuf, NA, 1);
    gemm(attbuf, P(bi + 4), P(bi + 5), xbuf, xbuf, Mg, Dm, Dm, 0);
    k_layernorm<<<(Mg + 7) / 8, 256, 0, stream>>>(xbuf, lnbuf, P(bi + 6), P(bi + 7), Mg);
    gemm(lnbuf, P(bi + 8), P(bi + 9), nullptr, hidbuf, Mg, DFF, Dm, 1);
    gemm(hidbuf, P(bi + 10), P(bi + 11), xbuf, xbuf, Mg, Dm, DFF, 0);
  }
  // 11) generator final LN -> g
  k_layernorm<<<(Mg + 7) / 8, 256, 0, stream>>>(xbuf, o_g, P(196), P(197), Mg);

  // 12) head: 384 -> 384(gelu) -> 640
  gemm(o_g, P(198), P(199), nullptr, hidbuf, Mg, Dm, Dm, 1);
  gemm(hidbuf, P(200), P(201), nullptr, o_logits, Mg, AV, Dm, 0);
}
